// LSTMAutoencoder_50672024158853
// MI455X (gfx1250) — compile-verified, matrix operand order checked by reading, not run on hardware
//
#include <hip/hip_runtime.h>

typedef _Float16 f16;
typedef __attribute__((ext_vector_type(16))) _Float16 v16h;
typedef __attribute__((ext_vector_type(8)))  _Float16 v8h;
typedef __attribute__((ext_vector_type(8)))  float    v8f;

constexpr int kT = 20, kF = 5, kH = 32, kBN = 16;
constexpr int kWaves = 8, kRowsPerWave = 16, kRowsPerBlock = kWaves * kRowsPerWave;
constexpr int kThreads = 256;

union AFrag { v16h v; v8h h[2]; };

// Fast sigmoid/tanh: v_exp_f32 + v_rcp_f32 (no IEEE divide chains).
__device__ __forceinline__ float fsig(float x) {
  return __builtin_amdgcn_rcpf(1.f + __expf(-x));
}
__device__ __forceinline__ float ftanh(float x) {
  return 1.f - 2.f * __builtin_amdgcn_rcpf(__expf(2.f * x) + 1.f);
}
// LDS ops from one wave execute in order; this waits for outstanding DS ops and
// is also a compiler memory barrier so stores/loads around the LDS transpose
// scratch cannot be reordered.
__device__ __forceinline__ void lds_fence() { asm volatile("s_wait_dscnt 0" ::: "memory"); }

__device__ __forceinline__ v16h zero_v16h() {
  v16h z;
#pragma unroll
  for (int e = 0; e < 16; ++e) z[e] = (f16)0.f;
  return z;
}

// B fragments are stored in LDS in per-lane element order: 32 bytes per lane,
// contiguous -> two ds_load_b128 per fragment.
__device__ __forceinline__ v16h load_frag(const f16* base, int tile, int lane) {
  AFrag a;
  const f16* p = base + tile * 512 + lane * 16;
  a.h[0] = *(const v8h*)(p);
  a.h[1] = *(const v8h*)(p + 8);
  return a.v;
}

// WMMA 16-bit operand lane/element -> K mapping (ISA 7.12.2):
// lane = hi*16 + n ; element e<8 -> K = hi*8+e ; e>=8 -> K = 16+hi*8+(e-8).
// B element (k, n) of tile holds W[tile*16+n][k]  (z = in @ W^T).
__device__ __forceinline__ void fill_frag(f16* dst, const float* W, int rowsN,
                                          int colsK, int ntiles, int tid) {
  const int total = ntiles * 512;
  for (int idx = tid; idx < total; idx += kThreads) {
    const int tile = idx >> 9, rem = idx & 511, lane = rem >> 4, e = rem & 15;
    const int n = lane & 15, hi = lane >> 4;
    const int k = (e < 8) ? (hi * 8 + e) : (16 + hi * 8 + (e - 8));
    const int row = tile * 16 + n;
    float v = 0.f;
    if (k < colsK && row < rowsN) v = W[row * colsK + k];
    dst[idx] = (f16)v;
  }
}

// One LSTM step for a 16-row batch tile held by one wave.
// a_in, a_h: 16x32 f16 A fragments. cs: cell state in C-layout (2 subtiles x 8).
// Returns new hidden state as an A fragment (via LDS transpose scratch).
__device__ __forceinline__ v16h lstm_step(v16h a_in, v16h a_h, v8f cs[2],
                                          const f16* wihf, const f16* whhf,
                                          const float* bias, f16* scr, int lane) {
  const int n = lane & 15, hi = lane >> 4;
  v8f acc[8];
#pragma unroll
  for (int t = 0; t < 8; ++t) {
    v8f c;
#pragma unroll
    for (int v = 0; v < 8; ++v) c[v] = bias[t];
    v16h bw = load_frag(wihf, t, lane);
    c = __builtin_amdgcn_wmma_f32_16x16x32_f16(false, a_in, false, bw, (short)0, c, false, false);
    v16h bh = load_frag(whhf, t, lane);
    c = __builtin_amdgcn_wmma_f32_16x16x32_f16(false, a_h, false, bh, (short)0, c, false, false);
    acc[t] = c;
  }
  // gates: z columns [0,32)=i [32,64)=f [64,96)=g [96,128)=o  -> tiles 2g+jj
#pragma unroll
  for (int jj = 0; jj < 2; ++jj) {
#pragma unroll
    for (int v = 0; v < 8; ++v) {
      const float iv = fsig(acc[0 + jj][v]);
      const float fv = fsig(acc[2 + jj][v]);
      const float gv = ftanh(acc[4 + jj][v]);
      const float ov = fsig(acc[6 + jj][v]);
      const float cc = fv * cs[jj][v] + iv * gv;
      cs[jj][v] = cc;
      const float hh = ov * ftanh(cc);
      // C-layout (m = v+hi*8, col = jj*16+n) -> row-major scratch [m][k]
      scr[(v + hi * 8) * 32 + jj * 16 + n] = (f16)hh;
    }
  }
  lds_fence();
  AFrag a;  // A-layout read: lane's row m = n, contiguous K runs of 8
  a.h[0] = *(const v8h*)(scr + n * 32 + hi * 8);
  a.h[1] = *(const v8h*)(scr + n * 32 + 16 + hi * 8);
  lds_fence();
  return a.v;
}

__global__ __launch_bounds__(kThreads) void lstm_ae_fused(
    const float* __restrict__ x,
    const float* __restrict__ eW0, const float* __restrict__ eU0,
    const float* __restrict__ eb0i, const float* __restrict__ eb0h,
    const float* __restrict__ eW1, const float* __restrict__ eU1,
    const float* __restrict__ eb1i, const float* __restrict__ eb1h,
    const float* __restrict__ bnW, const float* __restrict__ bnb,
    const float* __restrict__ exW, const float* __restrict__ exb,
    const float* __restrict__ dW0, const float* __restrict__ dU0,
    const float* __restrict__ db0i, const float* __restrict__ db0h,
    const float* __restrict__ dW1, const float* __restrict__ dU1,
    const float* __restrict__ db1i, const float* __restrict__ db1h,
    const float* __restrict__ oW, const float* __restrict__ obv_,
    float* __restrict__ out, int Btot) {
  __shared__ alignas(16) f16 s_wih[4][4096];   // 4 layers x 8 tiles (L0 K-padded)
  __shared__ alignas(16) f16 s_whh[4][4096];
  __shared__ alignas(16) f16 s_bnw[512];
  __shared__ alignas(16) f16 s_exw[1024];
  __shared__ alignas(16) f16 s_outw[512];
  __shared__ alignas(16) f16 s_scr[kWaves][512];  // per-wave transpose scratch

  const int tid = threadIdx.x;
  fill_frag(s_wih[0], eW0, 4 * kH, kF, 8, tid);
  fill_frag(s_whh[0], eU0, 4 * kH, kH, 8, tid);
  fill_frag(s_wih[1], eW1, 4 * kH, kH, 8, tid);
  fill_frag(s_whh[1], eU1, 4 * kH, kH, 8, tid);
  fill_frag(s_wih[2], dW0, 4 * kH, kH, 8, tid);
  fill_frag(s_whh[2], dU0, 4 * kH, kH, 8, tid);
  fill_frag(s_wih[3], dW1, 4 * kH, kH, 8, tid);
  fill_frag(s_whh[3], dU1, 4 * kH, kH, 8, tid);
  fill_frag(s_bnw, bnW, kBN, kH, 1, tid);
  fill_frag(s_exw, exW, kH, kBN, 2, tid);
  fill_frag(s_outw, oW, kF, kH, 1, tid);
  __syncthreads();

  const int lane = tid & 31;
  const int wave = tid >> 5;
  const int n = lane & 15, hi = lane >> 4;
  const long b0 = (long)blockIdx.x * kRowsPerBlock + (long)wave * kRowsPerWave;
  f16* scr = s_scr[wave];

  // ---------------- encoder (2 layers, interleaved per timestep) -------------
  // Load encoder biases only for this phase (keeps their liveness short).
  float be0[8], be1[8];
#pragma unroll
  for (int t = 0; t < 8; ++t) {
    be0[t] = eb0i[16 * t + n] + eb0h[16 * t + n];
    be1[t] = eb1i[16 * t + n] + eb1h[16 * t + n];
  }

  v16h A_h0 = zero_v16h(), A_h1 = zero_v16h();
  v8f cs0[2], cs1[2];
#pragma unroll
  for (int jj = 0; jj < 2; ++jj)
#pragma unroll
    for (int v = 0; v < 8; ++v) { cs0[jj][v] = 0.f; cs1[jj][v] = 0.f; }

#pragma unroll 1
  for (int t = 0; t < kT; ++t) {
    // build x A-fragment: only lanes hi==0 carry K=0..4 (rest zero-padded)
    v16h ax = zero_v16h();
    if (hi == 0) {
      const float* xp = x + ((b0 + n) * kT + t) * kF;
#pragma unroll
      for (int e = 0; e < kF; ++e) ax[e] = (f16)xp[e];
    }
    A_h0 = lstm_step(ax, A_h0, cs0, s_wih[0], s_whh[0], be0, scr, lane);
    A_h1 = lstm_step(A_h0, A_h1, cs1, s_wih[1], s_whh[1], be1, scr, lane);
  }

  // ---------------- bottleneck: relu(h @ bnW^T + b), relu(. @ exW^T + b) -----
  v8f accL;
  {
    const float bv = bnb[n];
#pragma unroll
    for (int v = 0; v < 8; ++v) accL[v] = bv;
    v16h bf = load_frag(s_bnw, 0, lane);
    accL = __builtin_amdgcn_wmma_f32_16x16x32_f16(false, A_h1, false, bf, (short)0, accL, false, false);
  }
  // zero scratch (latent is only 16 wide; K=16..31 of the A frag must be 0)
#pragma unroll
  for (int e = 0; e < 16; ++e) scr[lane * 16 + e] = (f16)0.f;
  lds_fence();
#pragma unroll
  for (int v = 0; v < 8; ++v) scr[(v + hi * 8) * 32 + n] = (f16)fmaxf(accL[v], 0.f);
  lds_fence();
  v16h A_lat;
  {
    AFrag a;
    a.h[0] = *(const v8h*)(scr + n * 32 + hi * 8);
    a.h[1] = *(const v8h*)(scr + n * 32 + 16 + hi * 8);
    lds_fence();
    A_lat = a.v;
  }
  v8f accE[2];
#pragma unroll
  for (int tt = 0; tt < 2; ++tt) {
    const float ev = exb[16 * tt + n];
#pragma unroll
    for (int v = 0; v < 8; ++v) accE[tt][v] = ev;
    v16h ef = load_frag(s_exw, tt, lane);
    accE[tt] = __builtin_amdgcn_wmma_f32_16x16x32_f16(false, A_lat, false, ef, (short)0, accE[tt], false, false);
  }
#pragma unroll
  for (int jj = 0; jj < 2; ++jj)
#pragma unroll
    for (int v = 0; v < 8; ++v)
      scr[(v + hi * 8) * 32 + jj * 16 + n] = (f16)fmaxf(accE[jj][v], 0.f);
  lds_fence();
  v16h A_e;
  {
    AFrag a;
    a.h[0] = *(const v8h*)(scr + n * 32 + hi * 8);
    a.h[1] = *(const v8h*)(scr + n * 32 + 16 + hi * 8);
    lds_fence();
    A_e = a.v;
  }

  // ---------------- decoder + fused output projection ------------------------
  float bd0[8], bd1[8];
#pragma unroll
  for (int t = 0; t < 8; ++t) {
    bd0[t] = db0i[16 * t + n] + db0h[16 * t + n];
    bd1[t] = db1i[16 * t + n] + db1h[16 * t + n];
  }

  v16h A_d0 = zero_v16h(), A_d1 = zero_v16h();
  v8f cd0[2], cd1[2];
#pragma unroll
  for (int jj = 0; jj < 2; ++jj)
#pragma unroll
    for (int v = 0; v < 8; ++v) { cd0[jj][v] = 0.f; cd1[jj][v] = 0.f; }
  const float obv = (n < kF) ? obv_[n] : 0.f;

#pragma unroll 1
  for (int t = 0; t < kT; ++t) {
    A_d0 = lstm_step(A_e, A_d0, cd0, s_wih[2], s_whh[2], bd0, scr, lane);
    A_d1 = lstm_step(A_d0, A_d1, cd1, s_wih[3], s_whh[3], bd1, scr, lane);
    v8f accO;
#pragma unroll
    for (int v = 0; v < 8; ++v) accO[v] = 0.f;
    v16h of = load_frag(s_outw, 0, lane);
    accO = __builtin_amdgcn_wmma_f32_16x16x32_f16(false, A_d1, false, of, (short)0, accO, false, false);
    if (n < kF) {
#pragma unroll
      for (int v = 0; v < 8; ++v)
        out[((b0 + v + hi * 8) * kT + t) * kF + n] = accO[v] + obv;
    }
  }
}

extern "C" void kernel_launch(void* const* d_in, const int* in_sizes, int n_in,
                              void* d_out, int out_size, void* d_ws, size_t ws_size,
                              hipStream_t stream) {
  (void)n_in; (void)out_size; (void)d_ws; (void)ws_size;
  const float* p[23];
  for (int i = 0; i < 23; ++i) p[i] = (const float*)d_in[i];
  const int Btot = in_sizes[0] / (kT * kF);
  const int blocks = (Btot + kRowsPerBlock - 1) / kRowsPerBlock;
  hipLaunchKernelGGL(lstm_ae_fused, dim3(blocks), dim3(kThreads), 0, stream,
                     p[0], p[1], p[2], p[3], p[4], p[5], p[6], p[7], p[8],
                     p[9], p[10], p[11], p[12], p[13], p[14], p[15], p[16],
                     p[17], p[18], p[19], p[20], p[21], p[22],
                     (float*)d_out, Btot);
}
